// FusionSlot_35725537968192
// MI455X (gfx1250) — matrix operand order repaired and checked by reading, not compiled, not run on hardware
//
#include <hip/hip_runtime.h>
#include <hip/hip_bf16.h>
#include <math.h>

typedef __attribute__((ext_vector_type(16))) _Float16 v16h;
typedef __attribute__((ext_vector_type(8)))  _Float16 h8;
typedef __attribute__((ext_vector_type(8)))  float    v8f;

#define D_MODEL 48
#define NSLOT   21
#define ROWSPB  16     // batch rows per block
#define AST     72     // f16 activation staging stride (halves), 144B = 16B aligned
#define WST     64     // packed weight row stride (halves), cols 48..63 zero
#define ITERS   3

// ---------------- WMMA helpers (CDNA5 wave32, 16x16x32 f16 -> f32) ----------------

__device__ __forceinline__ v8f wmma_f16(v16h a, v16h b, v8f c) {
  // (neg_a, A, neg_b, B, c_mod, C, reuse_a, reuse_b)
  return __builtin_amdgcn_wmma_f32_16x16x32_f16(false, a, false, b, (short)0, c,
                                                false, false);
}

__device__ __forceinline__ v16h pack16(h8 lo, h8 hi) {
  v16h r;
#pragma unroll
  for (int i = 0; i < 8; ++i) { r[i] = lo[i]; r[i + 8] = hi[i]; }
  return r;
}

// A fragment: 16(M) x 32(K) f16 from LDS staging, row stride AST halves.
// Lane layout (ISA 7.12.2): lanes 0-15 hold K = kb+[0..7] and kb+16+[0..7];
// lanes 16-31 hold K = kb+8+[0..7] and kb+24+[0..7].
__device__ __forceinline__ v16h load_a(const _Float16* st, int lane, int kb) {
  int m = lane & 15, sel = lane >> 4;
  const _Float16* p = st + m * AST + kb + sel * 8;
  return pack16(*(const h8*)p, *(const h8*)(p + 16));
}

// B fragment: 32(K) x 16(N): B[k][n] = W[row0+n][kb+k] from packed f16 weights
// ([out][WST], padded). Lane n = lane&15; lanes 0-15 K=kb+[0..7],kb+16+[0..7]; 16-31 K=kb+8+[0..7],kb+24+[0..7].
__device__ __forceinline__ v16h load_b(const _Float16* __restrict__ W, int lane,
                                       int row0, int kb) {
  int n = lane & 15, sel = lane >> 4;
  const _Float16* p = W + (size_t)(row0 + n) * WST + kb + sel * 8;
  return pack16(*(const h8*)p, *(const h8*)(p + 16));
}

__device__ __forceinline__ float sigmoidf_(float x) {
  return 1.0f / (1.0f + __expf(-x));
}
__device__ __forceinline__ float softplusf_(float x) {
  return log1pf(__expf(x));
}

// ---------------- weight packing: f32 [rows][48] -> f16 [rows][64] (zero pad) ------

extern "C" __global__ void fs_pack_w(const float* __restrict__ src,
                                     _Float16* __restrict__ dst, int rows) {
  int t = blockIdx.x * blockDim.x + threadIdx.x;
  if (t >= rows * WST) return;
  int r = t >> 6, c = t & 63;
  dst[t] = (c < D_MODEL) ? (_Float16)src[r * D_MODEL + c] : (_Float16)0.0f;
}

// ---------------- main fused kernel: 16 rows per 128-thread block ------------------

extern "C" __global__ void __launch_bounds__(128)
fs_fusion_kernel(const float* __restrict__ slot,   // [M,21,48]
                 const float* __restrict__ fq,     // [48]
                 const float* __restrict__ ipb,    // [144] in_proj_b (bq|bk|bv)
                 const float* __restrict__ opb,    // [48]
                 const float* __restrict__ lnqg, const float* __restrict__ lnqb,
                 const float* __restrict__ lnkg, const float* __restrict__ lnkb,
                 const float* __restrict__ gbi,    // [144] gru_b_ih
                 const float* __restrict__ gbh,    // [144] gru_b_hh
                 const float* __restrict__ syr,    // [48]
                 const float* __restrict__ pr,     // [1]
                 const float* __restrict__ p1b, const float* __restrict__ p2b,
                 const _Float16* __restrict__ WhIn,   // [144,64] (wq|wk|wv)
                 const _Float16* __restrict__ WhOut,  // [48,64]
                 const _Float16* __restrict__ WhGih,  // [144,64]
                 const _Float16* __restrict__ WhGhh,  // [144,64]
                 const _Float16* __restrict__ WhP1,   // [48,64]
                 const _Float16* __restrict__ WhP2,   // [48,64]
                 float* __restrict__ outF,            // [M,48]
                 float* __restrict__ outA,            // [M,21]
                 int M) {
  // staging roles: [0]=qn / fused, [1]=raw q, [2]=out / hidden, [3]=attn
  __shared__ __align__(16) _Float16 stage[4][ROWSPB * AST];
  __shared__ __align__(16) _Float16 Kl[ROWSPB * NSLOT * D_MODEL];  // f16 K tiles
  __shared__ __align__(16) _Float16 Vl[ROWSPB * NSLOT * D_MODEL];  // f16 V tiles
  __shared__ float qbuf[ROWSPB * D_MODEL];   // fp32 recurrent state q
  __shared__ float Qs[ROWSPB * 52];          // fp32 Q (stride 52: conflict pad)
  __shared__ float giS[ROWSPB * 148];        // fp32 gi (stride 148)
  __shared__ float ghS[ROWSPB * 148];        // fp32 gh
  __shared__ float sc[ROWSPB * 64];          // scores / attn weights [m][h*32+n]

  const int t = threadIdx.x;
  const int wave = t >> 5, lane = t & 31;
  const int m0 = blockIdx.x * ROWSPB;
  const int n16 = lane & 15, rb = (lane >> 4) * 8;
  const float scale = 0.20412414523193154f;  // 1/sqrt(24)

  // ---- init: zero staging pad cols [48..72) (read by kb=32 A-fragments), q = fq
  for (int i = t; i < 4 * ROWSPB * 24; i += 128) {
    int buf = i / (ROWSPB * 24), rem = i % (ROWSPB * 24);
    stage[buf][(rem / 24) * AST + 48 + (rem % 24)] = (_Float16)0.0f;
  }
  for (int i = t; i < ROWSPB * D_MODEL; i += 128) qbuf[i] = fq[i % D_MODEL];
  __syncthreads();

  // ---- Phase 0 (iteration-invariant): LN(kv), K = kvn@wk.T+bk, V = kvn@wv.T+bv
  // 336 kv rows = 21 WMMA M-tiles of 16; each chunk is 3KB contiguous in global.
  for (int c = wave; c < NSLOT; c += 4) {
    int r = lane & 15, hf = lane >> 4;  // lane -> (row-in-chunk, 24-col half)
    const float* src = slot + (size_t)m0 * (NSLOT * D_MODEL) + c * 768 + r * 48 + hf * 24;
    float4 xv[6];
#pragma unroll
    for (int i = 0; i < 6; ++i) xv[i] = *(const float4*)(src + 4 * i);
    const float* x = (const float*)xv;
    float s = 0.f, s2 = 0.f;
#pragma unroll
    for (int i = 0; i < 24; ++i) { s += x[i]; s2 += x[i] * x[i]; }
    s  += __shfl_xor(s, 16, 32);
    s2 += __shfl_xor(s2, 16, 32);
    float mu = s * (1.0f / 48.0f);
    float rstd = rsqrtf(s2 * (1.0f / 48.0f) - mu * mu + 1e-5f);
    _Float16* stg = stage[wave];
#pragma unroll
    for (int i = 0; i < 24; ++i) {
      int col = hf * 24 + i;
      stg[r * AST + col] = (_Float16)((x[i] - mu) * rstd * lnkg[col] + lnkb[col]);
    }
    // chained K-dim (48 -> 2 chunks of 32, zero padded), 3 N-tiles each for K and V
    v8f aK[3], aV[3];
#pragma unroll
    for (int tl = 0; tl < 3; ++tl) { aK[tl] = (v8f)(0.0f); aV[tl] = (v8f)(0.0f); }
#pragma unroll
    for (int kb = 0; kb <= 32; kb += 32) {
      v16h a = load_a(stg, lane, kb);
#pragma unroll
      for (int tl = 0; tl < 3; ++tl) {
        aK[tl] = wmma_f16(a, load_b(WhIn, lane, 48 + tl * 16, kb), aK[tl]);
        aV[tl] = wmma_f16(a, load_b(WhIn, lane, 96 + tl * 16, kb), aV[tl]);
      }
    }
#pragma unroll
    for (int tl = 0; tl < 3; ++tl) {
      float bk = ipb[48 + tl * 16 + n16], bv = ipb[96 + tl * 16 + n16];
#pragma unroll
      for (int rr = 0; rr < 8; ++rr) {
        int row = c * 16 + rb + rr;
        Kl[row * D_MODEL + tl * 16 + n16] = (_Float16)(aK[tl][rr] + bk);
        Vl[row * D_MODEL + tl * 16 + n16] = (_Float16)(aV[tl][rr] + bv);
      }
    }
  }
  __syncthreads();

  // ---- 3 refinement iterations ----
  for (int it = 0; it < ITERS; ++it) {
    // LN(q) -> stage[0] (qn, f16) ; raw q -> stage[1] (f16 for gru_hh GEMM)
    {
      int m = t >> 3, sub = t & 7;  // 8 threads per row, contiguous lanes
      float v[6], s = 0.f, s2 = 0.f;
#pragma unroll
      for (int k = 0; k < 6; ++k) {
        v[k] = qbuf[m * D_MODEL + sub * 6 + k];
        s += v[k]; s2 += v[k] * v[k];
      }
#pragma unroll
      for (int off = 1; off < 8; off <<= 1) {
        s  += __shfl_xor(s, off, 32);
        s2 += __shfl_xor(s2, off, 32);
      }
      float mu = s * (1.0f / 48.0f);
      float rstd = rsqrtf(s2 * (1.0f / 48.0f) - mu * mu + 1e-5f);
#pragma unroll
      for (int k = 0; k < 6; ++k) {
        int col = sub * 6 + k;
        stage[0][m * AST + col] = (_Float16)((v[k] - mu) * rstd * lnqg[col] + lnqb[col]);
        stage[1][m * AST + col] = (_Float16)v[k];
      }
    }
    __syncthreads();

    // Q = qn@wq.T + bq (3 tiles) and gh = q@w_hh.T + b_hh (9 tiles): 12 tiles over 4 waves
    for (int tl = wave; tl < 12; tl += 4) {
      const _Float16 *Ast, *W; float* outp; const float* bias; int colb, stride;
      if (tl < 3) { Ast = stage[0]; W = WhIn;  colb = tl * 16;       stride = 52;  outp = Qs;  bias = ipb; }
      else        { Ast = stage[1]; W = WhGhh; colb = (tl - 3) * 16; stride = 148; outp = ghS; bias = gbh; }
      v8f acc = (v8f)(0.0f);
#pragma unroll
      for (int kb = 0; kb <= 32; kb += 32)
        acc = wmma_f16(load_a(Ast, lane, kb), load_b(W, lane, colb, kb), acc);
      float bv = bias[colb + n16];
#pragma unroll
      for (int rr = 0; rr < 8; ++rr)
        outp[(rb + rr) * stride + colb + n16] = acc[rr] + bv;
    }
    __syncthreads();

    // scores[m][h][n] = (Q[m,h,:] . K[m,n,h,:]) / sqrt(24)
    {
      int m = t >> 3, sub = t & 7;
      for (int idx = sub; idx < 2 * NSLOT; idx += 8) {
        int h = idx / NSLOT, n = idx % NSLOT;
        const float*    Qp = Qs + m * 52 + h * 24;
        const _Float16* Kp = Kl + (m * NSLOT + n) * D_MODEL + h * 24;
        float s = 0.f;
#pragma unroll
        for (int d = 0; d < 24; ++d) s += Qp[d] * (float)Kp[d];
        sc[m * 64 + h * 32 + n] = s * scale;
      }
    }
    __syncthreads();

    // softmax over n (21): one thread per (m,h)
    if (t < 32) {
      float* base = sc + (t >> 1) * 64 + (t & 1) * 32;
      float mx = -1e30f;
      for (int n = 0; n < NSLOT; ++n) mx = fmaxf(mx, base[n]);
      float ssum = 0.f;
      for (int n = 0; n < NSLOT; ++n) { float e = __expf(base[n] - mx); base[n] = e; ssum += e; }
      float inv = 1.0f / ssum;
      for (int n = 0; n < NSLOT; ++n) base[n] *= inv;
    }
    __syncthreads();

    // attn[m][d] = sum_n aw[m][h(d)][n] * V[m,n,d]  -> stage[3] f16
    {
      int m = t >> 3, sub = t & 7;
      for (int d = sub; d < D_MODEL; d += 8) {
        const float* awp = sc + m * 64 + ((d >= 24) ? 32 : 0);
        float a = 0.f;
#pragma unroll
        for (int n = 0; n < NSLOT; ++n)
          a += awp[n] * (float)Vl[(m * NSLOT + n) * D_MODEL + d];
        stage[3][m * AST + d] = (_Float16)a;
      }
    }
    __syncthreads();

    // out = attn@wo.T + bo -> stage[2] f16 (3 tiles)
    for (int tl = wave; tl < 3; tl += 4) {
      v8f acc = (v8f)(0.0f);
#pragma unroll
      for (int kb = 0; kb <= 32; kb += 32)
        acc = wmma_f16(load_a(stage[3], lane, kb), load_b(WhOut, lane, tl * 16, kb), acc);
      float bv = opb[tl * 16 + n16];
#pragma unroll
      for (int rr = 0; rr < 8; ++rr)
        stage[2][(rb + rr) * AST + tl * 16 + n16] = (_Float16)(acc[rr] + bv);
    }
    __syncthreads();

    // gi = out@w_ih.T + b_ih (9 tiles)
    for (int tl = wave; tl < 9; tl += 4) {
      v8f acc = (v8f)(0.0f);
#pragma unroll
      for (int kb = 0; kb <= 32; kb += 32)
        acc = wmma_f16(load_a(stage[2], lane, kb), load_b(WhGih, lane, tl * 16, kb), acc);
      float bv = gbi[tl * 16 + n16];
#pragma unroll
      for (int rr = 0; rr < 8; ++rr)
        giS[(rb + rr) * 148 + tl * 16 + n16] = acc[rr] + bv;
    }
    __syncthreads();

    // GRU cell elementwise: q = (1-z)*n + z*q
    for (int i = t; i < ROWSPB * D_MODEL; i += 128) {
      int m = i / D_MODEL, d = i % D_MODEL;
      float ir = giS[m * 148 + d],      hr = ghS[m * 148 + d];
      float iz = giS[m * 148 + 48 + d], hz = ghS[m * 148 + 48 + d];
      float in_ = giS[m * 148 + 96 + d], hn = ghS[m * 148 + 96 + d];
      float r = sigmoidf_(ir + hr);
      float z = sigmoidf_(iz + hz);
      float nn = tanhf(in_ + r * hn);
      qbuf[i] = (1.0f - z) * nn + z * qbuf[i];
    }
    __syncthreads();
  }

  // ---- YieldActivation (elementwise) -> stage[0] f16
  {
    float p = 1.5f + softplusf_(pr[0]);
    for (int i = t; i < ROWSPB * D_MODEL; i += 128) {
      int m = i / D_MODEL, d = i % D_MODEL;
      float f = qbuf[i];
      float sig = softplusf_(syr[d]) + 0.01f;
      float ratio = fminf(fabsf(f / sig), 15.0f);
      f = f / __powf(1.0f + __powf(ratio, p), 1.0f / p);
      stage[0][m * AST + d] = (_Float16)f;
    }
  }
  __syncthreads();

  // ---- proj1 + ReLU -> stage[2] f16
  for (int tl = wave; tl < 3; tl += 4) {
    v8f acc = (v8f)(0.0f);
#pragma unroll
    for (int kb = 0; kb <= 32; kb += 32)
      acc = wmma_f16(load_a(stage[0], lane, kb), load_b(WhP1, lane, tl * 16, kb), acc);
    float bv = p1b[tl * 16 + n16];
#pragma unroll
    for (int rr = 0; rr < 8; ++rr)
      stage[2][(rb + rr) * AST + tl * 16 + n16] = (_Float16)fmaxf(acc[rr] + bv, 0.0f);
  }
  __syncthreads();

  // ---- proj2 -> global fused output
  for (int tl = wave; tl < 3; tl += 4) {
    v8f acc = (v8f)(0.0f);
#pragma unroll
    for (int kb = 0; kb <= 32; kb += 32)
      acc = wmma_f16(load_a(stage[2], lane, kb), load_b(WhP2, lane, tl * 16, kb), acc);
    float bv = p2b[tl * 16 + n16];
#pragma unroll
    for (int rr = 0; rr < 8; ++rr)
      outF[(size_t)(m0 + rb + rr) * D_MODEL + tl * 16 + n16] = acc[rr] + bv;
  }

  // ---- attn_w = mean over heads of final attention weights
  for (int i = t; i < ROWSPB * NSLOT; i += 128) {
    int m = i / NSLOT, n = i % NSLOT;
    outA[(size_t)(m0 + m) * NSLOT + n] = 0.5f * (sc[m * 64 + n] + sc[m * 64 + 32 + n]);
  }
}

// ---------------- host launch ----------------

extern "C" void kernel_launch(void* const* d_in, const int* in_sizes, int n_in,
                              void* d_out, int out_size, void* d_ws, size_t ws_size,
                              hipStream_t stream) {
  const float* slot = (const float*)d_in[0];
  const float* fq   = (const float*)d_in[1];
  const float* ipw  = (const float*)d_in[2];
  const float* ipb  = (const float*)d_in[3];
  const float* opw  = (const float*)d_in[4];
  const float* opb  = (const float*)d_in[5];
  const float* lnqg = (const float*)d_in[6];
  const float* lnqb = (const float*)d_in[7];
  const float* lnkg = (const float*)d_in[8];
  const float* lnkb = (const float*)d_in[9];
  const float* gwih = (const float*)d_in[10];
  const float* gwhh = (const float*)d_in[11];
  const float* gbi  = (const float*)d_in[12];
  const float* gbh  = (const float*)d_in[13];
  const float* syr  = (const float*)d_in[14];
  const float* pr   = (const float*)d_in[15];
  const float* p1w  = (const float*)d_in[16];
  const float* p1b  = (const float*)d_in[17];
  const float* p2w  = (const float*)d_in[18];
  const float* p2b  = (const float*)d_in[19];

  const int M = in_sizes[0] / (NSLOT * D_MODEL);  // 65536

  // pack weights to f16 [out][64] in workspace (L2-resident thereafter)
  _Float16* wsh = (_Float16*)d_ws;
  _Float16* WhIn  = wsh;             // 144*64
  _Float16* WhOut = wsh + 9216;      // 48*64
  _Float16* WhGih = wsh + 12288;     // 144*64
  _Float16* WhGhh = wsh + 21504;     // 144*64
  _Float16* WhP1  = wsh + 30720;     // 48*64
  _Float16* WhP2  = wsh + 33792;     // 48*64

  auto pack = [&](const float* s, _Float16* d, int rows) {
    int n = rows * WST;
    fs_pack_w<<<(n + 255) / 256, 256, 0, stream>>>(s, d, rows);
  };
  pack(ipw,  WhIn, 144);
  pack(opw,  WhOut, 48);
  pack(gwih, WhGih, 144);
  pack(gwhh, WhGhh, 144);
  pack(p1w,  WhP1, 48);
  pack(p2w,  WhP2, 48);

  float* outF = (float*)d_out;
  float* outA = outF + (size_t)M * D_MODEL;

  dim3 grid(M / ROWSPB), block(128);
  fs_fusion_kernel<<<grid, block, 0, stream>>>(
      slot, fq, ipb, opb, lnqg, lnqb, lnkg, lnkb, gbi, gbh, syr, pr, p1b, p2b,
      WhIn, WhOut, WhGih, WhGhh, WhP1, WhP2, outF, outA, M);
}
